// ImageDDS_19774029431565
// MI455X (gfx1250) — matrix operand order repaired and checked by reading, not compiled, not run on hardware
//
#include <hip/hip_runtime.h>
#include <math.h>

// ---------------------------------------------------------------------------
// CDNA5 (gfx1250) implementation of the DeepDDS-style GNN forward pass.
// Dense GEMMs: v_wmma_f32_16x16x32_f16 with register-double-buffered
// global->LDS staging (f32 -> f16), k-contiguous LDS layouts for both A and B
// so fragment gathers compile to wide ds_load_2addr_b64.
// Graph scatter ops use wave32 shuffles + global atomics.
// ---------------------------------------------------------------------------

typedef __attribute__((ext_vector_type(16))) _Float16 v16h;
typedef __attribute__((ext_vector_type(8)))  float    v8f;

#define BM 64
#define BN 64
#define BKK 32
#define KPAD 4             // row stride = 36 halves (72 B, 8-byte aligned)
#define ORD_NEG_INF 0x007FFFFFu   // ordf(-inf)

__device__ __forceinline__ unsigned ordf(float f) {
    unsigned u = __float_as_uint(f);
    return (u & 0x80000000u) ? ~u : (u | 0x80000000u);
}
__device__ __forceinline__ float unordf(unsigned o) {
    unsigned u = (o & 0x80000000u) ? (o ^ 0x80000000u) : ~o;
    return __uint_as_float(u);
}
__device__ __forceinline__ float lrelu_f(float v) { return v >= 0.f ? v : 0.01f * v; }
__device__ __forceinline__ float zero_nonfinite(float v) {
    unsigned u = __float_as_uint(v);
    return (((u >> 23) & 0xffu) == 0xffu) ? 0.f : v;
}

union H2 { _Float16 h[2]; unsigned u; };
union H4 { _Float16 h[4]; uint2 u2; };

// ---------------------------------------------------------------------------
// WMMA GEMM: C[M,Nc] = epilogue(A[M,K] @ W[K,Nc] + bias)
// REQUIRES: M % 64 == 0, Nc % 64 == 0, K even (all call sites satisfy this).
// Block 256 threads = 8 waves, 64x64 tile, K-steps of 32, software-pipelined:
// global loads for tile i+1 issue while tile i computes from LDS.
// Wave w -> m-tile (w&3), n-tiles (w>>2)*2 and (w>>2)*2+1.
// A staging: 4 float2 per thread (rows of A).
// W staging: 4 coalesced dwords per thread (k-column transpose in registers),
//            one ds_store_b64 -> Bs[n][k] with k contiguous.
// ---------------------------------------------------------------------------
template <int ACT>
__global__ __launch_bounds__(256)
void wmma_gemm_kernel(const float* __restrict__ A, const float* __restrict__ W,
                      const float* __restrict__ bias,
                      const float* __restrict__ scale, const float* __restrict__ shift,
                      float bnmul, float* __restrict__ C,
                      int M, int K, int Nc)
{
    __shared__ _Float16 As[BM][BKK + KPAD];   // [m][k]
    __shared__ _Float16 Bs[BN][BKK + KPAD];   // [n][k]

    const int tid  = threadIdx.x;
    const int wave = tid >> 5;
    const int lane = tid & 31;
    const int bm   = blockIdx.y * BM;
    const int bn   = blockIdx.x * BN;
    const int tm   = wave & 3;
    const int tn0  = (wave >> 2) << 1;

    // B staging coordinates: thread owns n = tid&63, k-group kg = tid>>6 (4 k's)
    const int sn = tid & 63;
    const int skg = (tid >> 6) * 4;

    float2 areg[4];
    float  wreg[4];

    auto load_tiles = [&](int kk) {
        if (kk + BKK <= K) {   // full tile: unguarded wide loads
#pragma unroll
            for (int i = 0; i < 4; ++i) {
                int c = tid + i * 256;
                int row = c >> 4, col = (c & 15) * 2;
                areg[i] = *(const float2*)(A + (size_t)(bm + row) * K + kk + col);
            }
#pragma unroll
            for (int j = 0; j < 4; ++j)   // coalesced across lanes (consecutive n)
                wreg[j] = W[(size_t)(kk + skg + j) * Nc + bn + sn];
        } else {               // K tail (at most the last tile): guarded, zero-fill
#pragma unroll
            for (int i = 0; i < 4; ++i) {
                int c = tid + i * 256;
                int row = c >> 4, col = (c & 15) * 2;
                float x0 = (kk + col     < K) ? A[(size_t)(bm + row) * K + kk + col]     : 0.f;
                float x1 = (kk + col + 1 < K) ? A[(size_t)(bm + row) * K + kk + col + 1] : 0.f;
                areg[i].x = x0; areg[i].y = x1;
            }
#pragma unroll
            for (int j = 0; j < 4; ++j) {
                int k = kk + skg + j;
                wreg[j] = (k < K) ? W[(size_t)k * Nc + bn + sn] : 0.f;
            }
        }
    };
    auto store_tiles = [&]() {
#pragma unroll
        for (int i = 0; i < 4; ++i) {
            int c = tid + i * 256;
            int row = c >> 4, col = (c & 15) * 2;
            H2 t; t.h[0] = (_Float16)areg[i].x; t.h[1] = (_Float16)areg[i].y;
            *(unsigned*)&As[row][col] = t.u;
        }
        H4 t;
        t.h[0] = (_Float16)wreg[0]; t.h[1] = (_Float16)wreg[1];
        t.h[2] = (_Float16)wreg[2]; t.h[3] = (_Float16)wreg[3];
        *(uint2*)&Bs[sn][skg] = t.u2;
    };

    v8f acc0 = {0.f, 0.f, 0.f, 0.f, 0.f, 0.f, 0.f, 0.f};
    v8f acc1 = {0.f, 0.f, 0.f, 0.f, 0.f, 0.f, 0.f, 0.f};

    load_tiles(0);
    for (int kk = 0; kk < K; kk += BKK) {
        store_tiles();
        __syncthreads();
        if (kk + BKK < K) load_tiles(kk + BKK);   // prefetch overlaps the WMMAs

        // A fragment (16x32): lanes 0-15 K=0..15, lanes 16-31 K=16..31 (matches B)
        v16h af;
        {
            int mrow = tm * 16 + (lane & 15);
            int kb   = (lane >> 4) * 16;
#pragma unroll
            for (int h = 0; h < 16; ++h) {
                int k = kb + h;
                af[h] = As[mrow][k];
            }
        }
        // B fragments (32x16): lanes 0-15 K=0..15, lanes 16-31 K=16..31; N=lane&15.
        // Bs is [n][k] with k contiguous -> 16 consecutive halves per lane.
        v16h bf0, bf1;
        {
            int kb2   = (lane >> 4) * 16;
            int ncol0 = tn0 * 16 + (lane & 15);
            int ncol1 = ncol0 + 16;
#pragma unroll
            for (int h = 0; h < 16; ++h) {
                bf0[h] = Bs[ncol0][kb2 + h];
                bf1[h] = Bs[ncol1][kb2 + h];
            }
        }
        acc0 = __builtin_amdgcn_wmma_f32_16x16x32_f16(false, af, false, bf0,
                                                      (short)0, acc0, false, false);
        acc1 = __builtin_amdgcn_wmma_f32_16x16x32_f16(false, af, false, bf1,
                                                      (short)0, acc1, false, false);
        __syncthreads();
    }

    // Write back (C/D layout: N = lane&15, M = r + 8*(lane>>4))
    const int nl = lane & 15;
    const int mo = (lane >> 4) * 8;
    const bool has_ss = (scale != nullptr);
#pragma unroll
    for (int t = 0; t < 2; ++t) {
        int gn = bn + (tn0 + t) * 16 + nl;
        v8f acc = t ? acc1 : acc0;
        float bv = bias ? bias[gn] : 0.f;
        float sc = has_ss ? scale[gn] * bnmul : 1.f;
        float sh = has_ss ? shift[gn] : 0.f;
#pragma unroll
        for (int r = 0; r < 8; ++r) {
            int gm = bm + tm * 16 + mo + r;
            float v = acc[r] + bv;
            if (has_ss) v = v * sc + sh;
            if (ACT == 1) v = lrelu_f(v);
            C[(size_t)gm * Nc + gn] = v;
        }
    }
}

// ---------------------------------------------------------------------------
// Elementwise / fill kernels
// ---------------------------------------------------------------------------
__global__ void fill_u32_kernel(unsigned* p, unsigned v, int n) {
    int i = blockIdx.x * blockDim.x + threadIdx.x;
    if (i < n) p[i] = v;
}
__global__ void fill_f32_kernel(float* p, float v, int n) {
    int i = blockIdx.x * blockDim.x + threadIdx.x;
    if (i < n) p[i] = v;
}
__global__ void lrelu_kernel(float* p, int n) {
    int i = blockIdx.x * blockDim.x + threadIdx.x;
    if (i < n) p[i] = lrelu_f(p[i]);
}

// ---------------------------------------------------------------------------
// Attention (TransformerConv) pieces.  ei layout: ei[0..E-1]=src, ei[E..]=dst.
// ---------------------------------------------------------------------------
__global__ __launch_bounds__(256)
void edge_logits_kernel(const float* __restrict__ q, const float* __restrict__ k,
                        const int* __restrict__ ei, int E, int H, int D,
                        float* __restrict__ logits, unsigned* __restrict__ menc)
{
    int gw = blockIdx.x * (blockDim.x >> 5) + (threadIdx.x >> 5);
    int lane = threadIdx.x & 31;
    if (gw >= E * H) return;
    int e = gw / H, h = gw - e * H;
    int src = ei[e], dst = ei[E + e];
    const float* qp = q + (size_t)dst * H * D + (size_t)h * D;
    const float* kp = k + (size_t)src * H * D + (size_t)h * D;
    float acc = 0.f;
    for (int d = lane; d < D; d += 32) acc += qp[d] * kp[d];
#pragma unroll
    for (int o = 16; o; o >>= 1) acc += __shfl_xor(acc, o, 32);
    if (lane == 0) {
        float lg = acc / sqrtf((float)D);
        logits[(size_t)e * H + h] = lg;
        atomicMax(&menc[(size_t)dst * H + h], ordf(lg));
    }
}

__global__ void m_finalize_kernel(const unsigned* __restrict__ menc,
                                  float* __restrict__ m, int n) {
    int i = blockIdx.x * blockDim.x + threadIdx.x;
    if (i < n) m[i] = zero_nonfinite(unordf(menc[i]));
}

__global__ void edge_expsum_kernel(float* __restrict__ logits, const int* __restrict__ ei,
                                   const float* __restrict__ m, float* __restrict__ z,
                                   int E, int H)
{
    int i = blockIdx.x * blockDim.x + threadIdx.x;
    if (i >= E * H) return;
    int e = i / H, h = i - e * H;
    int dst = ei[E + e];
    float ex = expf(logits[i] - m[(size_t)dst * H + h]);
    logits[i] = ex;
    atomicAdd(&z[(size_t)dst * H + h], ex);
}

__global__ __launch_bounds__(256)
void edge_aggregate_kernel(const float* __restrict__ logits, const float* __restrict__ z,
                           const float* __restrict__ v, const int* __restrict__ ei,
                           float* __restrict__ out, int E, int H, int D)
{
    int gw = blockIdx.x * (blockDim.x >> 5) + (threadIdx.x >> 5);
    int lane = threadIdx.x & 31;
    if (gw >= E * H) return;
    int e = gw / H, h = gw - e * H;
    int src = ei[e], dst = ei[E + e];
    float alpha = logits[(size_t)e * H + h] / (z[(size_t)dst * H + h] + 1e-16f);
    const float* vp = v + (size_t)src * H * D + (size_t)h * D;
    float* op = out + (size_t)dst * H * D + (size_t)h * D;
    for (int d = lane; d < D; d += 32)
        atomicAdd(&op[d], alpha * vp[d]);
}

// ---------------------------------------------------------------------------
// Global max pool per graph
// ---------------------------------------------------------------------------
__global__ void pool_scatter_kernel(const float* __restrict__ h, const int* __restrict__ batch,
                                    unsigned* __restrict__ genc, int N, int C)
{
    int i = blockIdx.x * blockDim.x + threadIdx.x;
    if (i >= N * C) return;
    int node = i / C, c = i - node * C;
    atomicMax(&genc[(size_t)batch[node] * C + c], ordf(h[i]));
}
__global__ void pool_finalize_kernel(const unsigned* __restrict__ genc,
                                     float* __restrict__ g, int n) {
    int i = blockIdx.x * blockDim.x + threadIdx.x;
    if (i < n) g[i] = zero_nonfinite(unordf(genc[i]));
}

// ---------------------------------------------------------------------------
// Row-wise L2 normalize: out = in / max(||in||, 1e-12). Wave per row.
// ---------------------------------------------------------------------------
__global__ __launch_bounds__(256)
void l2norm_rows_kernel(const float* __restrict__ in, float* __restrict__ out, int R, int C)
{
    int row = blockIdx.x * (blockDim.x >> 5) + (threadIdx.x >> 5);
    int lane = threadIdx.x & 31;
    if (row >= R) return;
    const float* ip = in + (size_t)row * C;
    float ss = 0.f;
    for (int c = lane; c < C; c += 32) { float v = ip[c]; ss += v * v; }
#pragma unroll
    for (int o = 16; o; o >>= 1) ss += __shfl_xor(ss, o, 32);
    float inv = 1.f / fmaxf(sqrtf(ss), 1e-12f);
    float* op = out + (size_t)row * C;
    for (int c = lane; c < C; c += 32) op[c] = ip[c] * inv;
}

__global__ void concat_kernel(const float* __restrict__ a, const float* __restrict__ b,
                              const float* __restrict__ c, float* __restrict__ xc, int B)
{
    int i = blockIdx.x * blockDim.x + threadIdx.x;
    if (i >= B * 384) return;
    int row = i / 384, col = i - row * 384;
    float v;
    if (col < 128)      v = a[row * 128 + col];
    else if (col < 256) v = b[row * 128 + (col - 128)];
    else                v = c[row * 128 + (col - 256)];
    xc[i] = v;
}

// Final: out[b] = sigmoid(dot(h2[b,:256], wf3) + bf3)
__global__ __launch_bounds__(256)
void fc3_sigmoid_kernel(const float* __restrict__ h2, const float* __restrict__ wf3,
                        const float* __restrict__ bf3, float* __restrict__ out, int B, int C)
{
    int row = blockIdx.x * (blockDim.x >> 5) + (threadIdx.x >> 5);
    int lane = threadIdx.x & 31;
    if (row >= B) return;
    const float* hp = h2 + (size_t)row * C;
    float acc = 0.f;
    for (int c = lane; c < C; c += 32) acc += hp[c] * wf3[c];
#pragma unroll
    for (int o = 16; o; o >>= 1) acc += __shfl_xor(acc, o, 32);
    if (lane == 0) {
        float x = acc + bf3[0];
        out[row] = 1.f / (1.f + expf(-x));
    }
}

// ---------------------------------------------------------------------------
// Host-side helpers
// ---------------------------------------------------------------------------
static void gemm(const float* A, const float* W, const float* bias,
                 const float* scale, const float* shift, float bnmul,
                 float* C, int M, int K, int Nc, int act, hipStream_t s)
{
    dim3 grid((Nc + BN - 1) / BN, (M + BM - 1) / BM), blk(256);
    if (act)
        wmma_gemm_kernel<1><<<grid, blk, 0, s>>>(A, W, bias, scale, shift, bnmul, C, M, K, Nc);
    else
        wmma_gemm_kernel<0><<<grid, blk, 0, s>>>(A, W, bias, scale, shift, bnmul, C, M, K, Nc);
}

static void run_attention(const float* Q, const float* Kb, const float* V, float* O,
                          const int* ei, float* logits, unsigned* menc, float* m, float* z,
                          int N, int E, int H, int D, hipStream_t s)
{
    int nh = N * H, eh = E * H;
    fill_u32_kernel<<<(nh + 255) / 256, 256, 0, s>>>(menc, ORD_NEG_INF, nh);
    edge_logits_kernel<<<(eh + 7) / 8, 256, 0, s>>>(Q, Kb, ei, E, H, D, logits, menc);
    m_finalize_kernel<<<(nh + 255) / 256, 256, 0, s>>>(menc, m, nh);
    fill_f32_kernel<<<(nh + 255) / 256, 256, 0, s>>>(z, 0.f, nh);
    edge_expsum_kernel<<<(eh + 255) / 256, 256, 0, s>>>(logits, ei, m, z, E, H);
    edge_aggregate_kernel<<<(eh + 7) / 8, 256, 0, s>>>(logits, z, V, ei, O, E, H, D);
}

extern "C" void kernel_launch(void* const* d_in, const int* in_sizes, int n_in,
                              void* d_out, int out_size, void* d_ws, size_t ws_size,
                              hipStream_t stream)
{
    (void)n_in; (void)ws_size;
    // ---- inputs (setup_inputs dict order) ----
    const float* x1  = (const float*)d_in[0];
    const int*   ei1 = (const int*)d_in[1];
    const int*   bt1 = (const int*)d_in[2];
    const float* x2  = (const float*)d_in[3];
    const int*   ei2 = (const int*)d_in[4];
    const int*   bt2 = (const int*)d_in[5];
    const float* cell = (const float*)d_in[6];
    const float *wq1 = (const float*)d_in[7],  *bq1 = (const float*)d_in[8];
    const float *wk1 = (const float*)d_in[9],  *bk1 = (const float*)d_in[10];
    const float *wv1 = (const float*)d_in[11], *bv1 = (const float*)d_in[12];
    const float *ws1 = (const float*)d_in[13], *bs1 = (const float*)d_in[14];
    const float *wq2 = (const float*)d_in[15], *bq2 = (const float*)d_in[16];
    const float *wk2 = (const float*)d_in[17], *bk2 = (const float*)d_in[18];
    const float *wv2 = (const float*)d_in[19], *bv2 = (const float*)d_in[20];
    const float *ws2 = (const float*)d_in[21], *bs2 = (const float*)d_in[22];
    const float *wg1 = (const float*)d_in[23], *bg1 = (const float*)d_in[24];
    const float *wg2 = (const float*)d_in[25], *bg2 = (const float*)d_in[26];
    const float *wr1 = (const float*)d_in[27], *br1 = (const float*)d_in[28];
    const float *wr2 = (const float*)d_in[29], *br2 = (const float*)d_in[30];
    const float *wr3 = (const float*)d_in[31], *br3 = (const float*)d_in[32];
    const float *wf1 = (const float*)d_in[33], *bf1 = (const float*)d_in[34];
    const float *gbn1 = (const float*)d_in[35], *bbn1 = (const float*)d_in[36];
    const float *wf2 = (const float*)d_in[37], *bf2 = (const float*)d_in[38];
    const float *gbn2 = (const float*)d_in[39], *bbn2 = (const float*)d_in[40];
    const float *wf3 = (const float*)d_in[41], *bf3 = (const float*)d_in[42];
    float* out = (float*)d_out; (void)out_size;

    const int N = in_sizes[0] / 78;
    const int E = in_sizes[1] / 2;
    const int B = in_sizes[6] / 954;
    const float BNM = 1.0f / sqrtf(1.0f + 1e-5f);

    // ---- workspace carve-up ----
    char* wp = (char*)d_ws;
    auto carve = [&](size_t bytes) -> void* {
        void* r = (void*)wp;
        wp += (bytes + 255) & ~(size_t)255;
        return r;
    };
    float* Q    = (float*)carve((size_t)N * 1024 * 4);
    float* Kb   = (float*)carve((size_t)N * 1024 * 4);
    float* V    = (float*)carve((size_t)N * 1024 * 4);
    float* O2   = (float*)carve((size_t)N * 1024 * 4);
    float* H1   = (float*)carve((size_t)N * 256 * 4);
    float* logits = (float*)carve((size_t)E * 2 * 4);
    unsigned* menc = (unsigned*)carve((size_t)N * 2 * 4);
    float* m    = (float*)carve((size_t)N * 2 * 4);
    float* z    = (float*)carve((size_t)N * 2 * 4);
    unsigned* genc = (unsigned*)carve((size_t)B * 1024 * 4);
    float* g    = (float*)carve((size_t)B * 1024 * 4);
    float* g1   = (float*)carve((size_t)B * 512 * 4);
    float* x1g  = (float*)carve((size_t)B * 128 * 4);
    float* x2g  = (float*)carve((size_t)B * 128 * 4);
    float* cn   = (float*)carve((size_t)B * 954 * 4);
    float* c1   = (float*)carve((size_t)B * 512 * 4);
    float* c2   = (float*)carve((size_t)B * 256 * 4);
    float* cv   = (float*)carve((size_t)B * 128 * 4);
    float* xc   = (float*)carve((size_t)B * 384 * 4);
    float* hf1  = (float*)carve((size_t)B * 512 * 4);
    float* hf2  = (float*)carve((size_t)B * 256 * 4);

    // ---- one drug branch ----
    auto branch = [&](const float* x, const int* ei, const int* bt, float* gout) {
        // TransformerConv1: 78 -> 2x128 (concat 256)
        gemm(x, wq1, bq1, nullptr, nullptr, 1.f, Q,  N, 78, 256, 0, stream);
        gemm(x, wk1, bk1, nullptr, nullptr, 1.f, Kb, N, 78, 256, 0, stream);
        gemm(x, wv1, bv1, nullptr, nullptr, 1.f, V,  N, 78, 256, 0, stream);
        gemm(x, ws1, bs1, nullptr, nullptr, 1.f, H1, N, 78, 256, 0, stream);  // skip path
        run_attention(Q, Kb, V, H1, ei, logits, menc, m, z, N, E, 2, 128, stream);
        lrelu_kernel<<<((size_t)N * 256 + 255) / 256, 256, 0, stream>>>(H1, N * 256);

        // TransformerConv2: 256 -> 2x512 (concat 1024)
        gemm(H1, wq2, bq2, nullptr, nullptr, 1.f, Q,  N, 256, 1024, 0, stream);
        gemm(H1, wk2, bk2, nullptr, nullptr, 1.f, Kb, N, 256, 1024, 0, stream);
        gemm(H1, wv2, bv2, nullptr, nullptr, 1.f, V,  N, 256, 1024, 0, stream);
        gemm(H1, ws2, bs2, nullptr, nullptr, 1.f, O2, N, 256, 1024, 0, stream); // skip path
        run_attention(Q, Kb, V, O2, ei, logits, menc, m, z, N, E, 2, 512, stream);
        lrelu_kernel<<<((size_t)N * 1024 + 255) / 256, 256, 0, stream>>>(O2, N * 1024);

        // global max pool per graph
        fill_u32_kernel<<<(B * 1024 + 255) / 256, 256, 0, stream>>>(genc, ORD_NEG_INF, B * 1024);
        pool_scatter_kernel<<<((size_t)N * 1024 + 255) / 256, 256, 0, stream>>>(O2, bt, genc, N, 1024);
        pool_finalize_kernel<<<(B * 1024 + 255) / 256, 256, 0, stream>>>(genc, g, B * 1024);

        // graph head: 1024 -> 512 (lrelu) -> 128, then L2-normalize rows
        gemm(g,  wg1, bg1, nullptr, nullptr, 1.f, g1,   B, 1024, 512, 1, stream);
        gemm(g1, wg2, bg2, nullptr, nullptr, 1.f, gout, B, 512,  128, 0, stream);
        l2norm_rows_kernel<<<(B + 7) / 8, 256, 0, stream>>>(gout, gout, B, 128);
    };

    branch(x1, ei1, bt1, x1g);
    branch(x2, ei2, bt2, x2g);

    // ---- cell branch ----
    l2norm_rows_kernel<<<(B + 7) / 8, 256, 0, stream>>>(cell, cn, B, 954);
    gemm(cn, wr1, br1, nullptr, nullptr, 1.f, c1, B, 954, 512, 1, stream);
    gemm(c1, wr2, br2, nullptr, nullptr, 1.f, c2, B, 512, 256, 1, stream);
    gemm(c2, wr3, br3, nullptr, nullptr, 1.f, cv, B, 256, 128, 0, stream);

    // ---- fusion head ----
    concat_kernel<<<(B * 384 + 255) / 256, 256, 0, stream>>>(x1g, x2g, cv, xc, B);
    l2norm_rows_kernel<<<(B + 7) / 8, 256, 0, stream>>>(xc, xc, B, 384);
    gemm(xc,  wf1, bf1, gbn1, bbn1, BNM, hf1, B, 384, 512, 1, stream);
    gemm(hf1, wf2, bf2, gbn2, bbn2, BNM, hf2, B, 512, 256, 1, stream);
    fc3_sigmoid_kernel<<<(B + 7) / 8, 256, 0, stream>>>(hf2, wf3, bf3, out, B, 256);
}
